// HybridTFTAccounting_13889924235937
// MI455X (gfx1250) — matrix operand order repaired and checked by reading, not compiled, not run on hardware
//
#include <hip/hip_runtime.h>
#include <hip/hip_bf16.h>

typedef __attribute__((ext_vector_type(16))) _Float16 v16h;
typedef __attribute__((ext_vector_type(8)))  _Float16 v8h;
typedef __attribute__((ext_vector_type(8)))  float    v8f;

// ---------------------------------------------------------------- math helpers
__device__ __forceinline__ float sigmoidf_(float x) { return 1.f / (1.f + __expf(-x)); }
__device__ __forceinline__ float eluf_(float x)     { return x > 0.f ? x : __expf(x) - 1.f; }
__device__ __forceinline__ float softplusf_(float x){ return x > 20.f ? x : log1pf(__expf(x)); }
__device__ __forceinline__ float nan0_(float x) {
    if (isnan(x)) return 0.f;
    if (isinf(x)) return x > 0.f ? 3.4028234663852886e38f : -3.4028234663852886e38f;
    return x;
}

// ---------------------------------------------------------------- WMMA helpers
// D = A(16x32 f16) * B(32x16 f16) + C(16x16 f32)
__device__ __forceinline__ v8f wmma_f16(v16h a, v16h b, v8f c) {
    return __builtin_amdgcn_wmma_f32_16x16x32_f16(false, a, false, b, (short)0, c, false, false);
}

// A tile load from row-major f16 [16 x ld], K-chunk kk (32 wide).
// lane<16: h[0..15]=K kk+0..15 ; lane>=16: +16 on K base.
// Two contiguous 16B runs per lane -> 2x b128 loads.
__device__ __forceinline__ v16h load_A(const _Float16* A, int ld, int kk, int lane) {
    int row = lane & 15;
    int kb  = kk + ((lane & 16) ? 16 : 0);
    const _Float16* p = A + row * ld + kb;
    v8h lo = *(const v8h*)(p);
    v8h hi = *(const v8h*)(p + 8);
    return __builtin_shufflevector(lo, hi, 0, 1, 2, 3, 4, 5, 6, 7, 8, 9, 10, 11, 12, 13, 14, 15);
}

// B tile load from pre-swizzled storage: tile = 512 halves, lane's 16 halves contiguous.
__device__ __forceinline__ v16h load_B_swz(const _Float16* Bsw, int tile, int lane) {
    return *(const v16h*)(Bsw + (size_t)tile * 512 + lane * 16);
}

// C/D store to row-major f32 [16 x ld] at col offset nn.
__device__ __forceinline__ void store_C(float* C, int ld, int nn, int lane, v8f c) {
    int col = nn + (lane & 15);
    int rb  = (lane & 16) ? 8 : 0;
#pragma unroll
    for (int j = 0; j < 8; ++j) C[(rb + j) * ld + col] = c[j];
}

// decode destination index of a swizzled B buffer -> (k, n) of the source matrix
__device__ __forceinline__ void decode_swz(int idx, int ntiles, int& k, int& n) {
    int tile = idx >> 9, within = idx & 511;
    int lane = within >> 4, i = within & 15;
    int kt = tile / ntiles, nt = tile - kt * ntiles;
    int nin = lane & 15;
    int kin = i + ((lane & 16) ? 16 : 0);
    k = kt * 32 + kin;
    n = nt * 16 + nin;
}

// ---------------------------------------------------------------- weight pack (f32 -> f16, B-swizzled tile layout)
__global__ void pack_kernel(const float* __restrict__ vh_w1, const float* __restrict__ vh_w2,
                            const float* __restrict__ vf_w1, const float* __restrict__ vf_w2,
                            const float* __restrict__ enc_wx, const float* __restrict__ enc_wh,
                            const float* __restrict__ dec_wx, const float* __restrict__ dec_wh,
                            const float* __restrict__ wk, const float* __restrict__ wv,
                            _Float16* __restrict__ w1h, _Float16* __restrict__ w2h,
                            _Float16* __restrict__ w1f, _Float16* __restrict__ w2f,
                            _Float16* __restrict__ Wenc, _Float16* __restrict__ Wdec,
                            _Float16* __restrict__ wkh, _Float16* __restrict__ wvh) {
    int i = blockIdx.x * blockDim.x + threadIdx.x;
    int nthr = gridDim.x * blockDim.x;
    int k, n;
    for (int idx = i; idx < 160 * 128; idx += nthr) {          // vh_w1 160x128
        decode_swz(idx, 8, k, n);
        w1h[idx] = (_Float16)vh_w1[k * 128 + n];
    }
    for (int idx = i; idx < 128 * 32; idx += nthr) {           // vh_w2 128x32
        decode_swz(idx, 2, k, n);
        w2h[idx] = (_Float16)vh_w2[k * 32 + n];
    }
    for (int idx = i; idx < 160 * 128; idx += nthr) {          // vf_w1 144x128 -> [x16 | pad16 | ctx128]
        decode_swz(idx, 8, k, n);
        float v = (k < 16) ? vf_w1[k * 128 + n] : (k < 32 ? 0.f : vf_w1[(k - 16) * 128 + n]);
        w1f[idx] = (_Float16)v;
    }
    for (int idx = i; idx < 128 * 16; idx += nthr) {           // vf_w2 128x16
        decode_swz(idx, 1, k, n);
        w2f[idx] = (_Float16)vf_w2[k * 16 + n];
    }
    for (int idx = i; idx < 160 * 512; idx += nthr) {          // [enc_wx(32); enc_wh(128)] x 512
        decode_swz(idx, 32, k, n);
        Wenc[idx] = (_Float16)((k < 32) ? enc_wx[k * 512 + n] : enc_wh[(k - 32) * 512 + n]);
    }
    for (int idx = i; idx < 160 * 512; idx += nthr) {          // [dec_wx(16); pad16; dec_wh(128)] x 512
        decode_swz(idx, 32, k, n);
        float v = (k < 16) ? dec_wx[k * 512 + n] : (k < 32 ? 0.f : dec_wh[(k - 32) * 512 + n]);
        Wdec[idx] = (_Float16)v;
    }
    for (int idx = i; idx < 128 * 128; idx += nthr) {          // wk 128x128
        decode_swz(idx, 8, k, n);
        wkh[idx] = (_Float16)wk[k * 128 + n];
    }
    for (int idx = i; idx < 128 * 128; idx += nthr) {          // wv 128x128
        decode_swz(idx, 8, k, n);
        wvh[idx] = (_Float16)wv[k * 128 + n];
    }
}

// ---------------------------------------------------------------- static ctx:  elu([emb_t, emb_s, size] @ W + b)
__global__ __launch_bounds__(128) void ctx_kernel(const float* __restrict__ te, const float* __restrict__ se,
                                                  const int* __restrict__ tids, const int* __restrict__ sids,
                                                  const float* __restrict__ slt,
                                                  const float* __restrict__ sw, const float* __restrict__ sb,
                                                  float* __restrict__ ctx) {
    int b = blockIdx.x, d = threadIdx.x;
    int ti = tids[b], si = sids[b];
    float acc = sb[d];
#pragma unroll
    for (int j = 0; j < 16; ++j) acc += te[ti * 16 + j] * sw[j * 128 + d];
#pragma unroll
    for (int j = 0; j < 8; ++j)  acc += se[si * 8 + j] * sw[(16 + j) * 128 + d];
    acc += slt[b] * sw[24 * 128 + d];
    ctx[b * 128 + d] = eluf_(acc);
}

// ---------------------------------------------------------------- VSN: hsel = x * softmax(elu([x|ctx]W1+b1)W2+b2)
// 16 rows/block, 4 waves. GEMM1: K=160 (5 chunks), N=128 (2 tiles/wave). GEMM2: K=128, N=F.
__global__ __launch_bounds__(128) void vsn_kernel(const float* __restrict__ x, const float* __restrict__ ctx,
                                                  const _Float16* __restrict__ W1, const float* __restrict__ b1,
                                                  const _Float16* __restrict__ W2, const float* __restrict__ b2,
                                                  _Float16* __restrict__ out, int T, int F) {
    __shared__ __align__(16) _Float16 W1s[160 * 128];   // 40 KB (swizzled)
    __shared__ __align__(16) _Float16 W2s[128 * 32];    //  8 KB (swizzled)
    __shared__ __align__(16) _Float16 Ash[16 * 160];    //  5 KB
    __shared__ __align__(16) _Float16 H1s[16 * 128];    //  4 KB
    __shared__ float    Ss[16 * 32];                    //  2 KB
    const int tid = threadIdx.x, lane = tid & 31, wid = tid >> 5;
    const int row0 = blockIdx.x * 16;

    for (int i = tid; i < (160 * 128) / 8; i += 128) ((uint4*)W1s)[i] = ((const uint4*)W1)[i];
    for (int i = tid; i < (128 * F) / 8;   i += 128) ((uint4*)W2s)[i] = ((const uint4*)W2)[i];
    for (int i = tid; i < 16 * 160; i += 128) {
        int r = i / 160, c = i % 160;
        int row = row0 + r, b = row / T;
        float v;
        if (c < F)       v = x[(size_t)row * F + c];
        else if (c < 32) v = 0.f;
        else             v = ctx[b * 128 + (c - 32)];
        Ash[i] = (_Float16)v;
    }
    __syncthreads();

    {   // GEMM1 + ELU
        int nA = wid * 32;
        v8f a0, a1;
        float bv0 = b1[nA + (lane & 15)], bv1 = b1[nA + 16 + (lane & 15)];
#pragma unroll
        for (int j = 0; j < 8; ++j) { a0[j] = bv0; a1[j] = bv1; }
#pragma unroll
        for (int kt = 0; kt < 5; ++kt) {
            v16h a = load_A(Ash, 160, kt * 32, lane);
            a0 = wmma_f16(a, load_B_swz(W1s, kt * 8 + wid * 2, lane), a0);
            a1 = wmma_f16(a, load_B_swz(W1s, kt * 8 + wid * 2 + 1, lane), a1);
        }
        int col = nA + (lane & 15);
        int rb  = (lane & 16) ? 8 : 0;
#pragma unroll
        for (int j = 0; j < 8; ++j) {
            H1s[(rb + j) * 128 + col]      = (_Float16)eluf_(a0[j]);
            H1s[(rb + j) * 128 + col + 16] = (_Float16)eluf_(a1[j]);
        }
    }
    __syncthreads();

    int ntiles2 = F >> 4;                 // 2 (hist) or 1 (future)
    if (wid < ntiles2) {                  // wave-uniform predicate
        int nn = wid * 16;
        v8f acc;
        float bv = b2[nn + (lane & 15)];
#pragma unroll
        for (int j = 0; j < 8; ++j) acc[j] = bv;
#pragma unroll
        for (int kt = 0; kt < 4; ++kt)
            acc = wmma_f16(load_A(H1s, 128, kt * 32, lane), load_B_swz(W2s, kt * ntiles2 + wid, lane), acc);
        store_C(Ss, 32, nn, lane, acc);
    }
    __syncthreads();

    if (tid < 16) {                       // per-row softmax gating
        int r = tid, row = row0 + r;
        float m = -1e30f;
        for (int c = 0; c < F; ++c) m = fmaxf(m, Ss[r * 32 + c]);
        float s = 0.f;
        for (int c = 0; c < F; ++c) s += __expf(Ss[r * 32 + c] - m);
        float inv = 1.f / s;
        for (int c = 0; c < F; ++c) {
            float wgt = __expf(Ss[r * 32 + c] - m) * inv;
            out[(size_t)row * F + c] = (_Float16)(x[(size_t)row * F + c] * wgt);
        }
    }
}

// ---------------------------------------------------------------- fused LSTM: z = [x_t | h] @ Wc + b  (K=160, N=512)
// 16 batch rows/block, 16 waves; each wave owns 32 gate columns, weights VGPR-resident across all T steps.
__global__ __launch_bounds__(512) void lstm_kernel(const _Float16* __restrict__ xsel, const float* __restrict__ maskp,
                                                   const _Float16* __restrict__ Wc, const float* __restrict__ bias,
                                                   const float* __restrict__ h_init, const float* __restrict__ c_init,
                                                   _Float16* __restrict__ hout16, float* __restrict__ hout32,
                                                   float* __restrict__ hfin, float* __restrict__ cfin,
                                                   int T, int XW) {
    __shared__ __align__(16) _Float16 Ash[16][160];     //  5 KB  [x_t | h] f16
    __shared__ float    csh[16][128];                   //  8 KB  cell state
    __shared__ float    zsh[16][512];                   // 32 KB  gates
    const int tid = threadIdx.x, lane = tid & 31, wid = tid >> 5;
    const int b0 = blockIdx.x * 16;
    const int n0 = wid * 32;

    v16h Wr[5][2];                        // resident weight slice (160x32), swizzled global
#pragma unroll
    for (int kt = 0; kt < 5; ++kt)
#pragma unroll
        for (int nt = 0; nt < 2; ++nt)
            Wr[kt][nt] = load_B_swz(Wc, kt * 32 + (n0 >> 4) + nt, lane);

    v8f bini[2];
#pragma unroll
    for (int nt = 0; nt < 2; ++nt) {
        float bv = bias[n0 + nt * 16 + (lane & 15)];
#pragma unroll
        for (int j = 0; j < 8; ++j) bini[nt][j] = bv;
    }

    for (int i = tid; i < 16 * 128; i += 512) {
        int r = i >> 7, u = i & 127;
        Ash[r][32 + u] = (_Float16)(h_init ? h_init[(b0 + r) * 128 + u] : 0.f);
        csh[r][u]      = c_init ? c_init[(b0 + r) * 128 + u] : 0.f;
    }
    if (XW < 32) {
        for (int i = tid; i < 16 * (32 - XW); i += 512) {
            int r = i / (32 - XW), c = XW + i % (32 - XW);
            Ash[r][c] = (_Float16)0.f;
        }
    }
    __syncthreads();

    const int xv = XW >> 3;               // uint4 chunks per row (4 or 2)
    for (int t = 0; t < T; ++t) {
        for (int i = tid; i < 16 * xv; i += 512) {
            int r = i / xv, c8 = i % xv;
            ((uint4*)&Ash[r][0])[c8] =
                ((const uint4*)(xsel + ((size_t)(b0 + r) * T + t) * XW))[c8];
        }
        __syncthreads();

        v8f acc0 = bini[0], acc1 = bini[1];
#pragma unroll
        for (int kt = 0; kt < 5; ++kt) {
            v16h a = load_A(&Ash[0][0], 160, kt * 32, lane);
            acc0 = wmma_f16(a, Wr[kt][0], acc0);
            acc1 = wmma_f16(a, Wr[kt][1], acc1);
        }
        store_C(&zsh[0][0], 512, n0, lane, acc0);
        store_C(&zsh[0][0], 512, n0 + 16, lane, acc1);
        __syncthreads();

        for (int i = tid; i < 16 * 128; i += 512) {
            int r = i >> 7, u = i & 127;
            float zi = zsh[r][u], zf = zsh[r][128 + u], zg = zsh[r][256 + u], zo = zsh[r][384 + u];
            float c = csh[r][u];
            float hprev = (float)Ash[r][32 + u];
            float cn = sigmoidf_(zf) * c + sigmoidf_(zi) * tanhf(zg);
            float hn = sigmoidf_(zo) * tanhf(cn);
            if (maskp && !(maskp[(size_t)(b0 + r) * T + t] > 0.5f)) { cn = c; hn = hprev; }
            csh[r][u] = cn;
            Ash[r][32 + u] = (_Float16)hn;
            size_t oidx = ((size_t)(b0 + r) * T + t) * 128 + u;
            if (hout16) hout16[oidx] = (_Float16)hn;
            if (hout32) hout32[oidx] = hn;
        }
        __syncthreads();
    }
    if (hfin) {
        for (int i = tid; i < 16 * 128; i += 512) {
            int r = i >> 7, u = i & 127;
            hfin[(b0 + r) * 128 + u] = (float)Ash[r][32 + u];
            cfin[(b0 + r) * 128 + u] = csh[r][u];
        }
    }
}

// ---------------------------------------------------------------- K/V projection: (B*L,128) @ (128,128) x2, WMMA
__global__ __launch_bounds__(128) void kv_kernel(const _Float16* __restrict__ enc,
                                                 const _Float16* __restrict__ Wk, const _Float16* __restrict__ Wv,
                                                 const float* __restrict__ bk, const float* __restrict__ bv,
                                                 _Float16* __restrict__ kout, _Float16* __restrict__ vout) {
    __shared__ __align__(16) _Float16 Ash[16 * 128];
    const int tid = threadIdx.x, lane = tid & 31, wid = tid >> 5;
    const size_t m0 = (size_t)blockIdx.x * 16;
    for (int i = tid; i < (16 * 128) / 8; i += 128)
        ((uint4*)Ash)[i] = ((const uint4*)(enc + m0 * 128))[i];
    __syncthreads();

    int n0 = wid * 32;
    v8f ak0, ak1, av0, av1;
    {
        float k0 = bk[n0 + (lane & 15)], k1 = bk[n0 + 16 + (lane & 15)];
        float v0 = bv[n0 + (lane & 15)], v1 = bv[n0 + 16 + (lane & 15)];
#pragma unroll
        for (int j = 0; j < 8; ++j) { ak0[j] = k0; ak1[j] = k1; av0[j] = v0; av1[j] = v1; }
    }
#pragma unroll
    for (int kt = 0; kt < 4; ++kt) {
        v16h a = load_A(Ash, 128, kt * 32, lane);
        ak0 = wmma_f16(a, load_B_swz(Wk, kt * 8 + (n0 >> 4), lane), ak0);
        ak1 = wmma_f16(a, load_B_swz(Wk, kt * 8 + (n0 >> 4) + 1, lane), ak1);
        av0 = wmma_f16(a, load_B_swz(Wv, kt * 8 + (n0 >> 4), lane), av0);
        av1 = wmma_f16(a, load_B_swz(Wv, kt * 8 + (n0 >> 4) + 1, lane), av1);
    }
    int col = n0 + (lane & 15);
    int rb  = (lane & 16) ? 8 : 0;
#pragma unroll
    for (int j = 0; j < 8; ++j) {
        kout[(m0 + rb + j) * 128 + col]      = (_Float16)ak0[j];
        kout[(m0 + rb + j) * 128 + col + 16] = (_Float16)ak1[j];
        vout[(m0 + rb + j) * 128 + col]      = (_Float16)av0[j];
        vout[(m0 + rb + j) * 128 + col + 16] = (_Float16)av1[j];
    }
}

// ---------------------------------------------------------------- attention: one block per (batch, head)
__global__ __launch_bounds__(256) void attn_kernel(const float* __restrict__ dec,
                                                   const float* __restrict__ wq, const float* __restrict__ bq,
                                                   const _Float16* __restrict__ kbuf, const _Float16* __restrict__ vbuf,
                                                   const float* __restrict__ maskp,
                                                   float* __restrict__ obuf) {
    __shared__ float decs[8 * 128];
    __shared__ float qs[8 * 32];
    __shared__ float sc[8 * 512];
    const int b = blockIdx.x, h = blockIdx.y;
    const int tid = threadIdx.x, lane = tid & 31, wid = tid >> 5;

    for (int i = tid; i < 1024; i += 256) decs[i] = dec[(size_t)b * 1024 + i];
    __syncthreads();
    {   // q = dec @ wq + bq, scale folded (1/sqrt(32))
        int t = tid >> 5, k = lane;
        float acc = bq[h * 32 + k];
        for (int d = 0; d < 128; ++d) acc += decs[t * 128 + d] * wq[d * 128 + h * 32 + k];
        qs[t * 32 + k] = acc * 0.17677669529663687f;
    }
    __syncthreads();
#pragma unroll
    for (int sh = 0; sh < 2; ++sh) {
        int s = tid + sh * 256;
        const _Float16* kp = kbuf + ((size_t)b * 512 + s) * 128 + h * 32;
        v16h k0 = *(const v16h*)kp;
        v16h k1 = *(const v16h*)(kp + 16);
        float kv[32];
#pragma unroll
        for (int k = 0; k < 16; ++k) { kv[k] = (float)k0[k]; kv[16 + k] = (float)k1[k]; }
        bool m = maskp[(size_t)b * 512 + s] > 0.5f;
        for (int t = 0; t < 8; ++t) {
            float acc = 0.f;
#pragma unroll
            for (int k = 0; k < 32; ++k) acc += qs[t * 32 + k] * kv[k];
            sc[t * 512 + s] = m ? acc : -1e9f;
        }
    }
    __syncthreads();
    {   // softmax: wave w handles query row w
        int t = wid;
        float m = -1e30f;
        for (int i = lane; i < 512; i += 32) m = fmaxf(m, sc[t * 512 + i]);
#pragma unroll
        for (int o = 16; o > 0; o >>= 1) m = fmaxf(m, __shfl_xor(m, o, 32));
        float s = 0.f;
        for (int i = lane; i < 512; i += 32) { float e = __expf(sc[t * 512 + i] - m); sc[t * 512 + i] = e; s += e; }
#pragma unroll
        for (int o = 16; o > 0; o >>= 1) s += __shfl_xor(s, o, 32);
        float inv = 1.f / s;
        for (int i = lane; i < 512; i += 32) sc[t * 512 + i] *= inv;
    }
    __syncthreads();
    {   // o = a @ v
        int t = tid >> 5, k = lane;
        const _Float16* vp = vbuf + (size_t)b * 512 * 128 + h * 32 + k;
        float acc = 0.f;
        for (int s = 0; s < 512; ++s) acc += sc[t * 512 + s] * (float)vp[(size_t)s * 128];
        obuf[(((size_t)b * 8 + t) * 4 + h) * 32 + k] = acc;
    }
}

// ---------------------------------------------------------------- output proj + residual + LN + head + accounting scan
__global__ __launch_bounds__(256) void head_kernel(const float* __restrict__ obuf, const float* __restrict__ dec,
                                                   const float* __restrict__ wo, const float* __restrict__ bo,
                                                   const float* __restrict__ projw, const float* __restrict__ projb,
                                                   const float* __restrict__ lng, const float* __restrict__ lnb,
                                                   const float* __restrict__ headw, const float* __restrict__ headb,
                                                   const float* __restrict__ y0, const float* __restrict__ xf,
                                                   const float* __restrict__ pdays, float* __restrict__ outp) {
    __shared__ float osh[1024];
    __shared__ float a1[1024];
    __shared__ float zsh[1024];
    __shared__ float raws[8 * 21];
    __shared__ float th[8 * 7 * 3];
    const int b = blockIdx.x, tid = threadIdx.x, lane = tid & 31, wid = tid >> 5;

    for (int i = tid; i < 1024; i += 256) osh[i] = obuf[(size_t)b * 1024 + i];
    __syncthreads();
    for (int i = tid; i < 1024; i += 256) {        // attn = o @ wo + bo
        int t = i >> 7, d = i & 127;
        float acc = bo[d];
        for (int j = 0; j < 128; ++j) acc += osh[t * 128 + j] * wo[j * 128 + d];
        a1[i] = acc;
    }
    __syncthreads();
    for (int i = tid; i < 1024; i += 256) {        // z = dec + attn @ proj_w + proj_b
        int t = i >> 7, d = i & 127;
        float acc = projb[d];
        for (int j = 0; j < 128; ++j) acc += a1[t * 128 + j] * projw[j * 128 + d];
        zsh[i] = dec[(size_t)b * 1024 + i] + acc;
    }
    __syncthreads();
    {   // LayerNorm per time row (wave per row)
        int t = wid;
        float s = 0.f;
        for (int i = lane; i < 128; i += 32) s += zsh[t * 128 + i];
#pragma unroll
        for (int o = 16; o > 0; o >>= 1) s += __shfl_xor(s, o, 32);
        float mu = s * (1.f / 128.f);
        float v = 0.f;
        for (int i = lane; i < 128; i += 32) { float d = zsh[t * 128 + i] - mu; v += d * d; }
#pragma unroll
        for (int o = 16; o > 0; o >>= 1) v += __shfl_xor(v, o, 32);
        float rinv = rsqrtf(v * (1.f / 128.f) + 1e-6f);
        for (int i = lane; i < 128; i += 32)
            zsh[t * 128 + i] = (zsh[t * 128 + i] - mu) * rinv * lng[i] + lnb[i];
    }
    __syncthreads();
    if (tid < 168) {                               // head: 8 x 21
        int t = tid / 21, j = tid % 21;
        float acc = headb[j];
        for (int d = 0; d < 128; ++d) acc += zsh[t * 128 + d] * headw[d * 21 + j];
        raws[tid] = acc;
    }
    __syncthreads();
    if (tid < 56) {                                // quantiles + sigmoid constrain
        int t = tid / 7, pi = tid % 7;
        float r50 = raws[t * 21 + pi * 3 + 0];
        float r10 = r50 - softplusf_(raws[t * 21 + pi * 3 + 1]);
        float r90 = r50 + softplusf_(raws[t * 21 + pi * 3 + 2]);
        const float LO[7] = {5.f, 5.f, 5.f, 0.f, 0.f, 0.f, -0.2f};
        const float HI[7] = {120.f, 180.f, 120.f, 0.3f, 0.3f, 1.f, 0.2f};
        float lo = LO[pi], d = HI[pi] - lo;
        th[(t * 7 + pi) * 3 + 0] = lo + d * sigmoidf_(r10);
        th[(t * 7 + pi) * 3 + 1] = lo + d * sigmoidf_(r50);
        th[(t * 7 + pi) * 3 + 2] = lo + d * sigmoidf_(r90);
    }
    __syncthreads();
    if (tid < 3) {                                 // accounting scan per quantile
        int q = tid;
        float K = y0[b * 8 + 3], STD = y0[b * 8 + 5], LTD = y0[b * 8 + 6], E = y0[b * 8 + 7];
        for (int t = 0; t < 8; ++t) {
            float dso  = th[(t * 7 + 0) * 3 + q];
            float dio  = th[(t * 7 + 1) * 3 + q];
            float dpo  = th[(t * 7 + 2) * 3 + q];
            float capx = th[(t * 7 + 3) * 3 + q];
            float depr = th[(t * 7 + 4) * 3 + q];
            float stds = th[(t * 7 + 5) * 3 + q];
            float dgr  = th[(t * 7 + 6) * 3 + q];
            float S    = nan0_(xf[(b * 8 + t) * 6 + 0]);
            float COGS = nan0_(xf[(b * 8 + t) * 6 + 1]);
            float EI   = nan0_(xf[(b * 8 + t) * 6 + 3]);
            float NI   = nan0_(xf[(b * 8 + t) * 6 + 4]);
            float Div  = nan0_(xf[(b * 8 + t) * 6 + 5]);
            float pd   = pdays[b * 8 + t];
            float ARn  = dso * S / pd;
            float Invn = dio * COGS / pd;
            float APn  = dpo * COGS / pd;
            float Kn   = K + capx * S - depr * K;
            float Dn   = (STD + LTD) * (1.f + dgr);
            float STDn = stds * Dn, LTDn = (1.f - stds) * Dn;
            float En   = E + NI - Div + EI;
            float TLn  = APn + STDn + LTDn;
            float Cn   = TLn + En - (ARn + Invn + Kn);
            float TAn  = Cn + ARn + Invn + Kn;
            float Eimp = TAn - TLn;
            float vals[10] = {Cn, ARn, Invn, Kn, APn, STDn, LTDn, TAn, TLn, Eimp};
            for (int ci = 0; ci < 10; ++ci)
                outp[(((size_t)b * 8 + t) * 10 + ci) * 3 + q] = vals[ci];
            K = Kn; STD = STDn; LTD = LTDn; E = Eimp;
        }
    }
}

// ---------------------------------------------------------------- host launcher
extern "C" void kernel_launch(void* const* d_in, const int* in_sizes, int n_in,
                              void* d_out, int out_size, void* d_ws, size_t ws_size,
                              hipStream_t stream) {
    (void)in_sizes; (void)n_in; (void)out_size; (void)ws_size;
    const int B = 256, L = 512, Hh = 8, FH = 32, FF = 16;

    const float* hist_feats = (const float*)d_in[0];
    const float* hist_mask  = (const float*)d_in[1];
    const float* fut_feats  = (const float*)d_in[2];
    const float* y0         = (const float*)d_in[3];
    const float* x_future   = (const float*)d_in[4];
    const float* pdf        = (const float*)d_in[5];
    const int*   ticker_id  = (const int*)d_in[6];
    const int*   sector_id  = (const int*)d_in[7];
    const float* size_lta   = (const float*)d_in[8];
    const float* ticker_emb = (const float*)d_in[9];
    const float* sector_emb = (const float*)d_in[10];
    const float* static_w   = (const float*)d_in[11];
    const float* static_b   = (const float*)d_in[12];
    const float* vh_w1      = (const float*)d_in[13];
    const float* vh_b1      = (const float*)d_in[14];
    const float* vh_w2      = (const float*)d_in[15];
    const float* vh_b2      = (const float*)d_in[16];
    const float* vf_w1      = (const float*)d_in[17];
    const float* vf_b1      = (const float*)d_in[18];
    const float* vf_w2      = (const float*)d_in[19];
    const float* vf_b2      = (const float*)d_in[20];
    const float* enc_wx     = (const float*)d_in[21];
    const float* enc_wh     = (const float*)d_in[22];
    const float* enc_b      = (const float*)d_in[23];
    const float* dec_wx     = (const float*)d_in[24];
    const float* dec_wh     = (const float*)d_in[25];
    const float* dec_b      = (const float*)d_in[26];
    const float* wq         = (const float*)d_in[27];
    const float* bq         = (const float*)d_in[28];
    const float* wk         = (const float*)d_in[29];
    const float* bk         = (const float*)d_in[30];
    const float* wv         = (const float*)d_in[31];
    const float* bv         = (const float*)d_in[32];
    const float* wo         = (const float*)d_in[33];
    const float* bo         = (const float*)d_in[34];
    const float* proj_w     = (const float*)d_in[35];
    const float* proj_b     = (const float*)d_in[36];
    const float* ln_g       = (const float*)d_in[37];
    const float* ln_b       = (const float*)d_in[38];
    const float* head_w     = (const float*)d_in[39];
    const float* head_b     = (const float*)d_in[40];
    float* outp = (float*)d_out;

    // workspace carve-out (~108 MB)
    char* wsp = (char*)d_ws;
    size_t off = 0;
    auto nxt = [&](size_t bytes) -> void* {
        void* p = wsp + off;
        off = (off + bytes + 255) & ~(size_t)255;
        return p;
    };
    float* ctx   = (float*)nxt((size_t)B * 128 * 4);
    float* hN    = (float*)nxt((size_t)B * 128 * 4);
    float* cN    = (float*)nxt((size_t)B * 128 * 4);
    float* decb  = (float*)nxt((size_t)B * Hh * 128 * 4);
    float* obuf  = (float*)nxt((size_t)B * Hh * 128 * 4);
    _Float16* hsel = (_Float16*)nxt((size_t)B * L * FH * 2);
    _Float16* fsel = (_Float16*)nxt((size_t)B * Hh * FF * 2);
    _Float16* ench = (_Float16*)nxt((size_t)B * L * 128 * 2);
    _Float16* kbuf = (_Float16*)nxt((size_t)B * L * 128 * 2);
    _Float16* vbuf = (_Float16*)nxt((size_t)B * L * 128 * 2);
    _Float16* w1h  = (_Float16*)nxt(160 * 128 * 2);
    _Float16* w2h  = (_Float16*)nxt(128 * 32 * 2);
    _Float16* w1f  = (_Float16*)nxt(160 * 128 * 2);
    _Float16* w2f  = (_Float16*)nxt(128 * 16 * 2);
    _Float16* Wenc = (_Float16*)nxt(160 * 512 * 2);
    _Float16* Wdec = (_Float16*)nxt(160 * 512 * 2);
    _Float16* wkh  = (_Float16*)nxt(128 * 128 * 2);
    _Float16* wvh  = (_Float16*)nxt(128 * 128 * 2);

    pack_kernel<<<256, 256, 0, stream>>>(vh_w1, vh_w2, vf_w1, vf_w2, enc_wx, enc_wh,
                                         dec_wx, dec_wh, wk, wv,
                                         w1h, w2h, w1f, w2f, Wenc, Wdec, wkh, wvh);
    ctx_kernel<<<B, 128, 0, stream>>>(ticker_emb, sector_emb, ticker_id, sector_id,
                                      size_lta, static_w, static_b, ctx);
    vsn_kernel<<<(B * L) / 16, 128, 0, stream>>>(hist_feats, ctx, w1h, vh_b1, w2h, vh_b2, hsel, L, FH);
    vsn_kernel<<<(B * Hh) / 16, 128, 0, stream>>>(fut_feats, ctx, w1f, vf_b1, w2f, vf_b2, fsel, Hh, FF);
    lstm_kernel<<<B / 16, 512, 0, stream>>>(hsel, hist_mask, Wenc, enc_b,
                                            nullptr, nullptr, ench, nullptr, hN, cN, L, FH);
    lstm_kernel<<<B / 16, 512, 0, stream>>>(fsel, nullptr, Wdec, dec_b,
                                            hN, cN, nullptr, decb, nullptr, nullptr, Hh, FF);
    kv_kernel<<<(B * L) / 16, 128, 0, stream>>>(ench, wkh, wvh, bk, bv, kbuf, vbuf);
    attn_kernel<<<dim3(B, 4), 256, 0, stream>>>(decb, wq, bq, kbuf, vbuf, hist_mask, obuf);
    head_kernel<<<B, 256, 0, stream>>>(obuf, decb, wo, bo, proj_w, proj_b, ln_g, ln_b,
                                       head_w, head_b, y0, x_future, pdf, outp);
}
